// MioA_42623255445558
// MI455X (gfx1250) — compile-verified, matrix operand order checked by reading, not run on hardware
//
#include <hip/hip_runtime.h>
#include <hip/hip_bf16.h>
#include <math.h>
#include <stdint.h>

// Problem constants
#define B_   4
#define L_   2048
#define D_   256
#define N_   16
#define R_   16
#define BL_  (B_ * L_)
#define K80  80       // 64 cols from Wx, 16 cols from Wxb packed at [64,80)
#define NC_  32       // scan chunks
#define CT_  64       // chunk length (NC_*CT_ == L_)

typedef _Float16     v16h  __attribute__((ext_vector_type(16)));
typedef float        v8f   __attribute__((ext_vector_type(8)));
typedef unsigned int u32x4 __attribute__((ext_vector_type(4)));
typedef int          i32x4 __attribute__((ext_vector_type(4)));
typedef int          i32x8 __attribute__((ext_vector_type(8)));

static __device__ __forceinline__ float softplus_f(float v) {
    return (v > 20.f) ? v : log1pf(__expf(v));
}

// ---------------------------------------------------------------------------
// TDM: issue a 2D tensor_load_to_lds of a [rows x 256] f32 row-major tensor.
// D# built per CDNA5 ISA §8.3/8.4 (count=1, type=2, data_size=4B).
// ---------------------------------------------------------------------------
static __device__ __forceinline__ void tdm_load_2d(uint32_t lds_off,
                                                   const float* gptr, int rows) {
    uint64_t ga = (uint64_t)(uintptr_t)gptr;
    u32x4 g0;
    g0[0] = 1u;                                            // count=1, user mode
    g0[1] = lds_off;                                       // lds_addr
    g0[2] = (uint32_t)ga;                                  // global_addr[31:0]
    g0[3] = (uint32_t)((ga >> 32) & 0x01FFFFFFu) | (2u << 30); // addr[56:32] | type=2
    i32x8 g1;
    g1[0] = (int)(2u << 16);          // data_size = 2 (4 bytes), no mask/pad/iter
    g1[1] = (int)(256u << 16);        // tensor_dim0[15:0] at bits 63:48
    g1[2] = (int)((uint32_t)rows << 16); // dim0[31:16]=0 | tensor_dim1[15:0]
    g1[3] = (int)(256u << 16);        // tensor_dim1[31:16]=0 | tile_dim0 = 256
    g1[4] = rows;                     // tile_dim1 = rows, tile_dim2 = 0
    g1[5] = 256;                      // tensor_dim0_stride[31:0]
    g1[6] = 0;
    g1[7] = 0;
    i32x4 z4 = {0, 0, 0, 0};
#if __clang_major__ >= 23
    i32x8 z8 = {0, 0, 0, 0, 0, 0, 0, 0};
    __builtin_amdgcn_tensor_load_to_lds(g0, g1, z4, z4, z8, 0);
#else
    __builtin_amdgcn_tensor_load_to_lds(g0, g1, z4, z4, 0);
#endif
}

// Load a 16x32 f16 A-fragment from an f32 row-major matrix (ld = leading dim).
// ISA 7.12.2: lanes 0-15 hold M=lane (K-base 0), lanes 16-31 hold M=lane-16 (K-base 8).
static __device__ __forceinline__ v16h load_frag_a(const float* __restrict__ base,
                                                   int ld, int row0, int k0, int lane) {
    int m  = lane & 15;
    int kb = (lane >> 4) << 3;
    const float* rp = base + (size_t)(row0 + m) * ld + k0;
    v16h a;
#pragma unroll
    for (int v = 0; v < 8; ++v) {
        int kk = ((v < 4) ? (2 * v) : (16 + 2 * (v - 4))) + kb;
        a[2 * v]     = (_Float16)rp[kk];
        a[2 * v + 1] = (_Float16)rp[kk + 1];
    }
    return a;
}

// ---------------------------------------------------------------------------
// Kernel A: x_dbl[8192,80] = X[8192,256] @ [Wx ; Wxb]^T
// Weights staged once per block into LDS by the Tensor Data Mover; 16 waves
// per block each compute one 16x16 tile via v_wmma_f32_16x16x32_f16.
// ---------------------------------------------------------------------------
__global__ void proj_gemm(const float* __restrict__ x,
                          const float* __restrict__ Wx,
                          const float* __restrict__ Wxb,
                          float* __restrict__ xdbl) {
    __shared__ float wlds[K80 * D_];            // 80 KB
    int warp = threadIdx.x >> 5;
    int lane = threadIdx.x & 31;

    if (warp == 0) {                            // one wave issues both TDM ops
        uint32_t base = (uint32_t)(size_t)(void*)wlds;
        tdm_load_2d(base, Wx, 64);
        tdm_load_2d(base + 64u * D_ * sizeof(float), Wxb, 16);
        __builtin_amdgcn_s_wait_tensorcnt(0);
    }
    __syncthreads();

    int tile = blockIdx.x * 16 + warp;          // 2560 tiles total
    int rt = tile / 5, ct = tile % 5;
    int row0 = rt * 16, col0 = ct * 16;

    int kb  = (lane >> 4) << 3;
    int col = col0 + (lane & 15);
    const float* wrow = wlds + (size_t)col * D_;   // B[k][n] = W[col][k] from LDS

    v8f c = {};
    for (int k0 = 0; k0 < D_; k0 += 32) {
        v16h a = load_frag_a(x, D_, row0, k0, lane);
        v16h b;
#pragma unroll
        for (int v = 0; v < 8; ++v) {
            int kk = ((v < 4) ? (2 * v) : (16 + 2 * (v - 4))) + kb;
            b[2 * v]     = (_Float16)wrow[k0 + kk];
            b[2 * v + 1] = (_Float16)wrow[k0 + kk + 1];
        }
        c = __builtin_amdgcn_wmma_f32_16x16x32_f16(false, a, false, b,
                                                   (short)0, c, false, false);
    }
    int rbase = row0 + ((lane >> 4) << 3);
#pragma unroll
    for (int r = 0; r < 8; ++r)
        xdbl[(size_t)(rbase + r) * K80 + col] = c[r];
}

// ---------------------------------------------------------------------------
// Kernel B: delta = softplus(xdbl[:,0:16] @ Wdt^T + bdt), q = same for [:,64:80]
// ---------------------------------------------------------------------------
__global__ void delta_gemm(const float* __restrict__ xdbl,
                           const float* __restrict__ Wdt,
                           const float* __restrict__ bdt,
                           float* __restrict__ deltaBuf,
                           float* __restrict__ qBuf) {
    int wave = blockIdx.x * (blockDim.x >> 5) + (threadIdx.x >> 5);
    int lane = threadIdx.x & 31;
    int rt = wave >> 4, ct = wave & 15;
    int row0 = rt * 16, col0 = ct * 16;

    int m  = lane & 15;
    int kb = (lane >> 4) << 3;
    const float* r1 = xdbl + (size_t)(row0 + m) * K80;
    const float* r2 = xdbl + (size_t)(row0 + m) * K80 + 64;
    const float* wr = Wdt  + (size_t)(col0 + m) * R_;

    v16h a1, a2, bf;
#pragma unroll
    for (int v = 0; v < 8; ++v) {
        int kk = ((v < 4) ? (2 * v) : (16 + 2 * (v - 4))) + kb;
        bool in = (kk < 16);
        a1[2 * v]     = in ? (_Float16)r1[kk]     : (_Float16)0.f;
        a1[2 * v + 1] = in ? (_Float16)r1[kk + 1] : (_Float16)0.f;
        a2[2 * v]     = in ? (_Float16)r2[kk]     : (_Float16)0.f;
        a2[2 * v + 1] = in ? (_Float16)r2[kk + 1] : (_Float16)0.f;
        bf[2 * v]     = in ? (_Float16)wr[kk]     : (_Float16)0.f;
        bf[2 * v + 1] = in ? (_Float16)wr[kk + 1] : (_Float16)0.f;
    }
    v8f c1 = {}, c2 = {};
    c1 = __builtin_amdgcn_wmma_f32_16x16x32_f16(false, a1, false, bf, (short)0, c1, false, false);
    c2 = __builtin_amdgcn_wmma_f32_16x16x32_f16(false, a2, false, bf, (short)0, c2, false, false);

    int rbase = row0 + ((lane >> 4) << 3);
    int col   = col0 + (lane & 15);
    float bias = bdt[col];
#pragma unroll
    for (int r = 0; r < 8; ++r) {
        size_t o = (size_t)(rbase + r) * D_ + col;
        deltaBuf[o] = softplus_f(c1[r] + bias);
        qBuf[o]     = softplus_f(c2[r] + bias);
    }
}

// ---------------------------------------------------------------------------
// Kernel E: w_eff = Wadapt @ Wout  (collapses the two output GEMMs)
// ---------------------------------------------------------------------------
__global__ void weff_kernel(const float* __restrict__ Wout,
                            const float* __restrict__ Wadapt,
                            float* __restrict__ weff) {
    int d = threadIdx.x;
    float s = 0.f;
    for (int o = 0; o < D_; ++o) s += Wadapt[o] * Wout[(size_t)o * D_ + d];
    weff[d] = s;
}

// ---------------------------------------------------------------------------
// Chunked scan, pass 1: per (b,chunk,d) compute local end-state (h0=0) and
// propagator P[n] = prod(dA[n]) over the chunk.  32x more parallelism than a
// monolithic scan (1024 waves instead of 32).
// ---------------------------------------------------------------------------
__global__ void scan_local(const float* __restrict__ x,
                           const float* __restrict__ xdbl,
                           const float* __restrict__ deltaBuf,
                           const float* __restrict__ qBuf,
                           const float* __restrict__ A_log,
                           float* __restrict__ hend,
                           float* __restrict__ Pprod) {
    int c = blockIdx.x, b = blockIdx.y, d = threadIdx.x;
    float An[N_];
#pragma unroll
    for (int n = 0; n < N_; ++n) An[n] = -__expf(A_log[d * N_ + n]);
    float h[N_], P[N_];
#pragma unroll
    for (int n = 0; n < N_; ++n) { h[n] = 0.f; P[n] = 1.f; }

    for (int t = 0; t < CT_; ++t) {
        int l    = c * CT_ + t;
        int row  = b * L_ + l;
        int rowf = b * L_ + (L_ - 1 - l);
        float delta = deltaBuf[(size_t)row  * D_ + d];
        float xv    = x[(size_t)row  * D_ + d];
        float u     = x[(size_t)rowf * D_ + d];
        float db    = qBuf[(size_t)rowf * D_ + d];
        const float* xr = xdbl + (size_t)row * K80;
        float dx = delta * xv, du = db * u;
#pragma unroll
        for (int n = 0; n < N_; ++n) {
            float dA = __expf(delta * An[n]);
            h[n] = dA * h[n] + dx * xr[16 + n] + du * xr[32 + n];
            P[n] *= dA;
        }
    }
    size_t o = (((size_t)b * NC_ + c) * D_ + d) * N_;
#pragma unroll
    for (int n = 0; n < N_; ++n) { hend[o + n] = h[n]; Pprod[o + n] = P[n]; }
}

// Pass 2: tiny sequential chain across chunks -> true chunk-initial states.
__global__ void scan_chain(const float* __restrict__ hend,
                           const float* __restrict__ Pprod,
                           float* __restrict__ hinit) {
    int b = blockIdx.x, d = threadIdx.x;
    float h[N_];
#pragma unroll
    for (int n = 0; n < N_; ++n) h[n] = 0.f;
    for (int c = 0; c < NC_; ++c) {
        size_t o = (((size_t)b * NC_ + c) * D_ + d) * N_;
#pragma unroll
        for (int n = 0; n < N_; ++n) {
            hinit[o + n] = h[n];
            h[n] = Pprod[o + n] * h[n] + hend[o + n];
        }
    }
}

// Pass 3: re-run each chunk seeded with its true h_init; emit y + skip terms.
__global__ void scan_final(const float* __restrict__ x,
                           const float* __restrict__ xdbl,
                           const float* __restrict__ deltaBuf,
                           const float* __restrict__ qBuf,
                           const float* __restrict__ A_log,
                           const float* __restrict__ Dp,
                           const float* __restrict__ hinit,
                           float* __restrict__ ytot) {
    int c = blockIdx.x, b = blockIdx.y, d = threadIdx.x;
    float An[N_];
#pragma unroll
    for (int n = 0; n < N_; ++n) An[n] = -__expf(A_log[d * N_ + n]);
    float dpd = Dp[d];
    float h[N_];
    size_t o = (((size_t)b * NC_ + c) * D_ + d) * N_;
#pragma unroll
    for (int n = 0; n < N_; ++n) h[n] = hinit[o + n];

    for (int t = 0; t < CT_; ++t) {
        int l    = c * CT_ + t;
        int row  = b * L_ + l;
        int rowf = b * L_ + (L_ - 1 - l);
        float delta = deltaBuf[(size_t)row  * D_ + d];
        float xv    = x[(size_t)row  * D_ + d];
        float u     = x[(size_t)rowf * D_ + d];
        float db    = qBuf[(size_t)rowf * D_ + d];
        const float* xr = xdbl + (size_t)row * K80;
        float dx = delta * xv, du = db * u;
        float y = 0.f;
#pragma unroll
        for (int n = 0; n < N_; ++n) {
            float dA = __expf(delta * An[n]);
            h[n] = dA * h[n] + dx * xr[16 + n] + du * xr[32 + n];
            y += h[n] * xr[48 + n];
        }
        ytot[(size_t)row * D_ + d] = y + (xv + u) * dpd;
    }
}

// ---------------------------------------------------------------------------
// Kernel D: y[row] = dot(ytot[row,:], w_eff) -- one wave per row, shfl reduce.
// ---------------------------------------------------------------------------
__global__ void out_reduce(const float* __restrict__ ytot,
                           const float* __restrict__ weff,
                           float* __restrict__ yout) {
    int wave = blockIdx.x * (blockDim.x >> 5) + (threadIdx.x >> 5);
    int lane = threadIdx.x & 31;
    if (wave >= BL_) return;
    const float* row = ytot + (size_t)wave * D_;
    float s = 0.f;
#pragma unroll
    for (int i = 0; i < 8; ++i) s += row[lane + 32 * i] * weff[lane + 32 * i];
#pragma unroll
    for (int off = 16; off > 0; off >>= 1) s += __shfl_xor(s, off, 32);
    if (lane == 0) yout[wave] = s;
}

extern "C" void kernel_launch(void* const* d_in, const int* in_sizes, int n_in,
                              void* d_out, int out_size, void* d_ws, size_t ws_size,
                              hipStream_t stream) {
    (void)in_sizes; (void)n_in; (void)out_size; (void)ws_size;
    const float* x      = (const float*)d_in[0];
    const float* Wx     = (const float*)d_in[1];
    const float* Wxb    = (const float*)d_in[2];
    const float* Wdt    = (const float*)d_in[3];
    const float* bdt    = (const float*)d_in[4];
    const float* A_log  = (const float*)d_in[5];
    const float* Dp     = (const float*)d_in[6];
    const float* Wout   = (const float*)d_in[7];
    const float* Wadapt = (const float*)d_in[8];
    float* out = (float*)d_out;

    float* ws       = (float*)d_ws;
    float* xdbl     = ws;                                    // 8192*80
    float* deltaBuf = xdbl     + (size_t)BL_ * K80;          // 8192*256
    float* qBuf     = deltaBuf + (size_t)BL_ * D_;           // 8192*256
    float* ytot     = qBuf     + (size_t)BL_ * D_;           // 8192*256
    float* weff     = ytot     + (size_t)BL_ * D_;           // 256
    float* hend     = weff     + D_;                         // B*NC*D*N
    float* Pprod    = hend     + (size_t)B_ * NC_ * D_ * N_;
    float* hinit    = Pprod    + (size_t)B_ * NC_ * D_ * N_;

    proj_gemm <<<2560 / 16, 512, 0, stream>>>(x, Wx, Wxb, xdbl);
    delta_gemm<<<(512 * 16) / 4, 128, 0, stream>>>(xdbl, Wdt, bdt, deltaBuf, qBuf);
    weff_kernel<<<1, 256, 0, stream>>>(Wout, Wadapt, weff);

    dim3 sg(NC_, B_);
    scan_local<<<sg, 256, 0, stream>>>(x, xdbl, deltaBuf, qBuf, A_log, hend, Pprod);
    scan_chain<<<B_, 256, 0, stream>>>(hend, Pprod, hinit);
    scan_final<<<sg, 256, 0, stream>>>(x, xdbl, deltaBuf, qBuf, A_log, Dp, hinit, ytot);

    out_reduce<<<BL_ / 8, 256, 0, stream>>>(ytot, weff, out);

    // tuple output: (y, x) -> copy x verbatim after the 8192 y values
    hipMemcpyAsync(out + BL_, x, (size_t)BL_ * D_ * sizeof(float),
                   hipMemcpyDeviceToDevice, stream);
}
